// GraphEmbedding_11836929868237
// MI455X (gfx1250) — 16-bit and 8-bit matrix operand order repaired and checked by reading, compiled, not run on hardware
//
#include <hip/hip_runtime.h>
#include <hip/hip_bf16.h>
#include <math.h>

#define N_NODES 512
#define TOPK    153
#define EMB     64
#define FDIM    64
#define NBATCH  32
#define EPSF    1e-8f

typedef __attribute__((ext_vector_type(16))) _Float16 v16h;
typedef __attribute__((ext_vector_type(8)))  float    v8f;
typedef __attribute__((ext_vector_type(8)))  int      v8i;

union AF16 { unsigned int u[8]; v16h v; };

__global__ void k_norms(const float* __restrict__ emb, float* __restrict__ norms,
                        int* __restrict__ maxc) {
  int i = blockIdx.x * blockDim.x + threadIdx.x;
  if (i == 0) *maxc = 0;
  if (i < N_NODES) {
    float s = 0.f;
#pragma unroll
    for (int k = 0; k < EMB; ++k) { float e = emb[i * EMB + k]; s += e * e; }
    norms[i] = sqrtf(s);
  }
}

__global__ void k_cos(const float* __restrict__ emb, const float* __restrict__ norms,
                      float* __restrict__ cosm) {
  __shared__ float se[EMB];
  int i = blockIdx.x;
  for (int k = threadIdx.x; k < EMB; k += blockDim.x) se[k] = emb[i * EMB + k];
  __syncthreads();
  float ni = norms[i];
  for (int j = threadIdx.x; j < N_NODES; j += blockDim.x) {
    float d = 0.f;
#pragma unroll 8
    for (int k = 0; k < EMB; ++k) d += se[k] * emb[j * EMB + k];
    cosm[i * N_NODES + j] = d / (ni * norms[j] + EPSF);
  }
}

__global__ void k_topk(const float* __restrict__ cosm, unsigned char* __restrict__ adjdir) {
  __shared__ float sc[N_NODES];
  int i = blockIdx.x;
  for (int j = threadIdx.x; j < N_NODES; j += blockDim.x) sc[j] = cosm[i * N_NODES + j];
  __syncthreads();
  for (int j = threadIdx.x; j < N_NODES; j += blockDim.x) {
    float vj = sc[j];
    int rank = 0;
    for (int k = 0; k < N_NODES; ++k) {
      float vk = sc[k];
      rank += (vk > vj) || (vk == vj && k < j);
    }
    adjdir[i * N_NODES + j] = (rank < TOPK) ? 1 : 0;
  }
}

__global__ void k_nb(const unsigned char* __restrict__ adjdir, unsigned char* __restrict__ nb) {
  int idx = blockIdx.x * blockDim.x + threadIdx.x;
  int i = idx >> 9, j = idx & 511;
  unsigned char s = adjdir[i * N_NODES + j] | adjdir[j * N_NODES + i];
  nb[idx] = (unsigned char)(s | (unsigned char)(i == j));
}

__global__ void k_common(const unsigned char* __restrict__ nb, int* __restrict__ common,
                         int* __restrict__ maxc) {
  int t  = blockIdx.x;
  int tm = t >> 5, tn = t & 31;
  int l  = threadIdx.x;
  int hi = l >> 4;
  int mrow = tm * 16 + (l & 15);
  int nrow = tn * 16 + (l & 15);
  v8i acc = {0, 0, 0, 0, 0, 0, 0, 0};
  for (int kb = 0; kb < N_NODES / 64; ++kb) {
    v8i a, b;
#pragma unroll
    for (int v = 0; v < 8; ++v) {
      int ka = kb * 64 + (v >> 1) * 16 + hi * 8 + (v & 1) * 4;
      a[v] = *(const int*)(nb + mrow * N_NODES + ka);
      int kbyt = kb * 64 + (v >> 1) * 16 + hi * 8 + (v & 1) * 4;
      b[v] = *(const int*)(nb + nrow * N_NODES + kbyt);
    }
    acc = __builtin_amdgcn_wmma_i32_16x16x64_iu8(false, a, false, b, acc, false, false);
  }
  int nloc = l & 15;
  int lmax = 0;
#pragma unroll
  for (int r = 0; r < 8; ++r) {
    int m = tm * 16 + r + hi * 8;
    int n = tn * 16 + nloc;
    int c = acc[r];
    common[m * N_NODES + n] = c;
    lmax = (c > lmax) ? c : lmax;
  }
  atomicMax(maxc, lmax);
}

__global__ void k_deg(const unsigned char* __restrict__ adjdir, const int* __restrict__ common,
                      const int* __restrict__ maxc, float* __restrict__ dinv) {
  __shared__ float red[256];
  int i = blockIdx.x;
  float mc = (float)(*maxc);
  float s = 0.f;
  for (int j = threadIdx.x; j < N_NODES; j += blockDim.x) {
    int c = common[i * N_NODES + j];
    float cv = (c > 1) ? ((float)c * (float)c) / mc : 0.f;
    s += adjdir[i * N_NODES + j] ? cv : 0.f;
  }
  red[threadIdx.x] = s;
  __syncthreads();
  for (int o = 128; o > 0; o >>= 1) {
    if (threadIdx.x < o) red[threadIdx.x] += red[threadIdx.x + o];
    __syncthreads();
  }
  if (threadIdx.x == 0) {
    float d = red[0];
    dinv[i] = (d > 0.f) ? rsqrtf(d) : 0.f;
  }
}

__global__ void k_buildA(const unsigned char* __restrict__ adjdir, const int* __restrict__ common,
                         const int* __restrict__ maxc, const float* __restrict__ dinv,
                         _Float16* __restrict__ Ah) {
  int idx = blockIdx.x * blockDim.x + threadIdx.x;
  int i = idx >> 9, j = idx & 511;
  float mc = (float)(*maxc);
  int c = common[idx];
  float cv = (c > 1) ? ((float)c * (float)c) / mc : 0.f;
  float a = adjdir[idx] ? dinv[i] * dinv[j] * cv : 0.f;
  Ah[idx] = (_Float16)a;
}

__global__ void k_cvt_x(const float* __restrict__ x, _Float16* __restrict__ xh) {
  int idx = blockIdx.x * blockDim.x + threadIdx.x;
  xh[idx] = (_Float16)x[idx];
}
__global__ void k_cvt_w(const float* __restrict__ w, _Float16* __restrict__ wT) {
  int idx = blockIdx.x * blockDim.x + threadIdx.x;
  int k = idx >> 6, n = idx & 63;
  wT[n * 64 + k] = (_Float16)w[k * 64 + n];
}

__global__ void k_h(const _Float16* __restrict__ xh, const _Float16* __restrict__ wT,
                    _Float16* __restrict__ hT) {
  int t = blockIdx.x;
  int b = t >> 5; int mt = t & 31;
  int l = threadIdx.x; int hi = l >> 4; int nloc = l & 15;
  const _Float16* arow = xh + (size_t)(b * N_NODES + mt * 16 + (l & 15)) * FDIM;
  v8f z = {0.f, 0.f, 0.f, 0.f, 0.f, 0.f, 0.f, 0.f};
  v8f acc[4] = {z, z, z, z};
#pragma unroll
  for (int kb = 0; kb < 2; ++kb) {
    AF16 a;
#pragma unroll
    for (int v = 0; v < 8; ++v) {
      int k0 = kb * 32 + ((v >= 4) ? 16 : 0) + hi * 8 + (v & 3) * 2;
      a.u[v] = *(const unsigned int*)(arow + k0);
    }
#pragma unroll
    for (int nt = 0; nt < 4; ++nt) {
      AF16 bf;
      const _Float16* brow = wT + (size_t)(nt * 16 + nloc) * FDIM;
#pragma unroll
      for (int v = 0; v < 8; ++v) {
        int k0b = kb * 32 + ((v >= 4) ? 16 : 0) + hi * 8 + (v & 3) * 2;
        bf.u[v] = *(const unsigned int*)(brow + k0b);
      }
      acc[nt] = __builtin_amdgcn_wmma_f32_16x16x32_f16(false, a.v, false, bf.v,
                                                       (short)0, acc[nt], false, false);
    }
  }
  _Float16* hrow = hT + (size_t)b * FDIM * N_NODES;
#pragma unroll
  for (int nt = 0; nt < 4; ++nt)
#pragma unroll
    for (int r = 0; r < 8; ++r) {
      int m = mt * 16 + r + hi * 8;
      int n = nt * 16 + nloc;
      hrow[n * N_NODES + m] = (_Float16)acc[nt][r];
    }
}

__global__ void k_out(const _Float16* __restrict__ Ah, const _Float16* __restrict__ hT,
                      const float* __restrict__ bias, float* __restrict__ out) {
  __shared__ _Float16 As[16 * N_NODES];
  int mt = blockIdx.x & 31;
  int bg = blockIdx.x >> 5;
  int tid = threadIdx.x;

  {
    const _Float16* gsrc = Ah + (size_t)mt * 16 * N_NODES + tid * 8;
    unsigned ldsoff = (unsigned)(uintptr_t)(&As[tid * 8]);
#pragma unroll
    for (int it = 0; it < 4; ++it) {
      unsigned long long ga = (unsigned long long)(uintptr_t)gsrc + (unsigned long long)it * 4096ull;
      unsigned lo = ldsoff + (unsigned)it * 4096u;
      asm volatile("global_load_async_to_lds_b128 %0, %1, off"
                   :: "v"(lo), "v"(ga) : "memory");
    }
    asm volatile("s_wait_asynccnt 0x0" ::: "memory");
  }
  __syncthreads();

  int w = tid >> 5;
  int b = bg * 8 + w;
  int l = tid & 31; int hi = l >> 4; int nloc = l & 15;
  const _Float16* As_row = &As[(l & 15) * N_NODES];
  const _Float16* bbase = hT + (size_t)b * FDIM * N_NODES;

  v8f z = {0.f, 0.f, 0.f, 0.f, 0.f, 0.f, 0.f, 0.f};
  v8f acc[4] = {z, z, z, z};
  for (int kb = 0; kb < N_NODES / 32; ++kb) {
    AF16 a;
#pragma unroll
    for (int v = 0; v < 8; ++v) {
      int k0 = kb * 32 + ((v >= 4) ? 16 : 0) + hi * 8 + (v & 3) * 2;
      a.u[v] = *(const unsigned int*)(As_row + k0);
    }
#pragma unroll
    for (int nt = 0; nt < 4; ++nt) {
      AF16 bf;
      const _Float16* brow = bbase + (size_t)(nt * 16 + nloc) * N_NODES;
#pragma unroll
      for (int v = 0; v < 8; ++v) {
        int k0b = kb * 32 + ((v >= 4) ? 16 : 0) + hi * 8 + (v & 3) * 2;
        bf.u[v] = *(const unsigned int*)(brow + k0b);
      }
      acc[nt] = __builtin_amdgcn_wmma_f32_16x16x32_f16(false, a.v, false, bf.v,
                                                       (short)0, acc[nt], false, false);
    }
  }
#pragma unroll
  for (int nt = 0; nt < 4; ++nt)
#pragma unroll
    for (int r = 0; r < 8; ++r) {
      int m = mt * 16 + r + hi * 8;
      int n = nt * 16 + nloc;
      out[((size_t)b * N_NODES + m) * FDIM + n] = acc[nt][r] + bias[n];
    }
}

extern "C" void kernel_launch(void* const* d_in, const int* in_sizes, int n_in,
                              void* d_out, int out_size, void* d_ws, size_t ws_size,
                              hipStream_t stream) {
  const float* x      = (const float*)d_in[0];
  const float* weight = (const float*)d_in[1];
  const float* bias   = (const float*)d_in[2];
  const float* emb    = (const float*)d_in[3];
  float* out = (float*)d_out;

  char* p = (char*)d_ws;
  auto alloc = [&](size_t bytes) -> void* {
    void* r = (void*)p;
    p += (bytes + 255) & ~(size_t)255;
    return r;
  };
  float*         norms  = (float*)alloc(N_NODES * sizeof(float));
  float*         dinv   = (float*)alloc(N_NODES * sizeof(float));
  int*           maxc   = (int*)alloc(sizeof(int));
  float*         cosm   = (float*)alloc((size_t)N_NODES * N_NODES * sizeof(float));
  unsigned char* adjdir = (unsigned char*)alloc((size_t)N_NODES * N_NODES);
  unsigned char* nb     = (unsigned char*)alloc((size_t)N_NODES * N_NODES);
  int*           common = (int*)alloc((size_t)N_NODES * N_NODES * sizeof(int));
  _Float16*      Ah     = (_Float16*)alloc((size_t)N_NODES * N_NODES * sizeof(_Float16));
  _Float16*      xh     = (_Float16*)alloc((size_t)NBATCH * N_NODES * FDIM * sizeof(_Float16));
  _Float16*      wT     = (_Float16*)alloc((size_t)FDIM * FDIM * sizeof(_Float16));
  _Float16*      hT     = (_Float16*)alloc((size_t)NBATCH * FDIM * N_NODES * sizeof(_Float16));

  k_norms <<<2, 256, 0, stream>>>(emb, norms, maxc);
  k_cos   <<<N_NODES, 256, 0, stream>>>(emb, norms, cosm);
  k_topk  <<<N_NODES, 256, 0, stream>>>(cosm, adjdir);
  k_nb    <<<(N_NODES * N_NODES) / 256, 256, 0, stream>>>(adjdir, nb);
  k_common<<<(N_NODES / 16) * (N_NODES / 16), 32, 0, stream>>>(nb, common, maxc);
  k_deg   <<<N_NODES, 256, 0, stream>>>(adjdir, common, maxc, dinv);
  k_buildA<<<(N_NODES * N_NODES) / 256, 256, 0, stream>>>(adjdir, common, maxc, dinv, Ah);
  k_cvt_x <<<(NBATCH * N_NODES * FDIM) / 256, 256, 0, stream>>>(x, xh);
  k_cvt_w <<<(FDIM * FDIM) / 256, 256, 0, stream>>>(weight, wT);
  k_h     <<<NBATCH * (N_NODES / 16), 32, 0, stream>>>(xh, wT, hT);
  k_out   <<<(N_NODES / 16) * (NBATCH / 8), 256, 0, stream>>>(Ah, hT, bias, out);
  (void)in_sizes; (void)n_in; (void)out_size; (void)ws_size;
}
